// MultiHeadAttention_34110630265666
// MI455X (gfx1250) — compile-verified, matrix operand order checked by reading, not run on hardware
//
#include <hip/hip_runtime.h>

// ---- types ---------------------------------------------------------------
typedef __attribute__((ext_vector_type(8)))  float  v8f;
typedef __attribute__((ext_vector_type(16))) __bf16 v16bf;
typedef __attribute__((ext_vector_type(8)))  __bf16 v8bf;

// ---- problem constants ---------------------------------------------------
#define BB     4
#define SS     1024
#define DMODEL 1024
#define NH     16
#define DH     64
#define MROWS  (BB * SS)          // 4096 rows for the projection GEMMs

// workspace layout (bytes): each bf16 head tensor is B*H*S*D*2 = 8 MB
#define WS_Q   ((size_t)0)
#define WS_K   ((size_t)8  * 1024 * 1024)
#define WS_VT  ((size_t)16 * 1024 * 1024)
#define WS_CTX ((size_t)24 * 1024 * 1024)

// output layout (floats)
#define OUT_X      ((size_t)0)
#define OUT_PRESK  ((size_t)4194304)
#define OUT_PRESV  ((size_t)8388608)
#define OUT_ATTN   ((size_t)12582912)

// Build a 16x16x32 bf16 WMMA operand fragment per the CDNA5 ISA layout:
// lane holds row/col (lane&15); K chunks at (lane>>4)*8 and (lane>>4)*8+16.
__device__ __forceinline__ v16bf make_frag(const __bf16* p) {
  v8bf lo = *reinterpret_cast<const v8bf*>(p);
  v8bf hi = *reinterpret_cast<const v8bf*>(p + 16);
  v16bf r;
#pragma unroll
  for (int i = 0; i < 8; ++i) { r[i] = lo[i]; r[8 + i] = hi[i]; }
  return r;
}

__device__ __forceinline__ v8f zero8() {
  v8f z;
#pragma unroll
  for (int i = 0; i < 8; ++i) z[i] = 0.0f;
  return z;
}

// ---- GEMM: C[M,N] = A[M,K] * W[N,K]^T + bias  (bf16 WMMA, f32 accum) -----
// mode 0: Q proj  -> dst_bf = Qh  [B,H,S,D] bf16
// mode 1: K proj  -> dst_bf = Kh  [B,H,S,D] bf16, dst_f32 = present_k f32
// mode 2: V proj  -> dst_bf = Vt  [B,H,D,S] bf16, dst_f32 = present_v f32 ([B,H,S,D])
// mode 3: O proj  -> A is bf16 ctx, dst_f32 = x [B,S,DM] f32
__global__ __launch_bounds__(128)
void mha_gemm_kernel(const void* __restrict__ Av, const float* __restrict__ W,
                     const float* __restrict__ bias, int mode,
                     __bf16* __restrict__ dst_bf, float* __restrict__ dst_f32) {
  __shared__ __align__(16) __bf16 As[64 * 32];
  __shared__ __align__(16) __bf16 Bs[64 * 32];

  const int t    = threadIdx.x;
  const int wave = t >> 5;
  const int lane = t & 31;
  const int kh   = lane >> 4;     // which K-half a lane owns
  const int mr   = lane & 15;     // row/col within 16
  const int blockN = blockIdx.x * 64;
  const int blockM = blockIdx.y * 64;

  const float*  Af = (const float*)Av;
  const __bf16* Ab = (const __bf16*)Av;

  v8f acc[4];
#pragma unroll
  for (int i = 0; i < 4; ++i) acc[i] = zero8();

  const int ldrow = t >> 1;           // 0..63
  const int ldcb  = (t & 1) * 16;     // 0 or 16

  for (int kb = 0; kb < DMODEL; kb += 32) {
    if (mode == 3) {
      const __bf16* src = Ab + (size_t)(blockM + ldrow) * DMODEL + kb + ldcb;
#pragma unroll
      for (int i = 0; i < 16; ++i) As[ldrow * 32 + ldcb + i] = src[i];
    } else {
      const float* src = Af + (size_t)(blockM + ldrow) * DMODEL + kb + ldcb;
#pragma unroll
      for (int i = 0; i < 16; ++i) As[ldrow * 32 + ldcb + i] = (__bf16)src[i];
    }
    {
      const float* src = W + (size_t)(blockN + ldrow) * DMODEL + kb + ldcb;
#pragma unroll
      for (int i = 0; i < 16; ++i) Bs[ldrow * 32 + ldcb + i] = (__bf16)src[i];
    }
    __syncthreads();

    v16bf a = make_frag(&As[(wave * 16 + mr) * 32 + kh * 8]);
#pragma unroll
    for (int nt = 0; nt < 4; ++nt) {
      v16bf b = make_frag(&Bs[(nt * 16 + mr) * 32 + kh * 8]);
      acc[nt] = __builtin_amdgcn_wmma_f32_16x16x32_bf16(
          false, a, false, b, (short)0, acc[nt], false, false);
    }
    __syncthreads();
  }

#pragma unroll
  for (int nt = 0; nt < 4; ++nt) {
    const int n  = blockN + nt * 16 + mr;
    const float bv = bias[n];
#pragma unroll
    for (int r = 0; r < 8; ++r) {
      const int m = blockM + wave * 16 + r + 8 * kh;
      const float val = acc[nt][r] + bv;
      if (mode == 3) {
        dst_f32[(size_t)m * DMODEL + n] = val;
      } else {
        const int bi = m >> 10, s = m & 1023;
        const int h  = n >> 6,  d = n & 63;
        const size_t ibhsd = (((size_t)(bi * NH + h)) * SS + s) * DH + d;
        if (mode == 2) {
          dst_bf[(((size_t)(bi * NH + h)) * DH + d) * SS + s] = (__bf16)val; // V^T
          dst_f32[ibhsd] = val;                                             // present_v
        } else {
          dst_bf[ibhsd] = (__bf16)val;                                      // Qh / Kh
          if (mode == 1) dst_f32[ibhsd] = val;                              // present_k
        }
      }
    }
  }
}

// ---- attention: one WG (4 waves) per (b, h, 16-row q tile) ---------------
__global__ __launch_bounds__(128)
void mha_attn_kernel(const __bf16* __restrict__ Qh, const __bf16* __restrict__ Kh,
                     const __bf16* __restrict__ Vt, float* __restrict__ attn,
                     __bf16* __restrict__ ctx) {
  extern __shared__ char smem[];
  float*  sc = (float*)smem;                      // [16][1024] scores, 64 KB
  __bf16* P  = (__bf16*)(smem + 16 * SS * 4);     // [16][1024] probs,  32 KB

  const int t    = threadIdx.x;
  const int wave = t >> 5;
  const int lane = t & 31;
  const int kh   = lane >> 4;
  const int mr   = lane & 15;
  const int qt = blockIdx.x, h = blockIdx.y, b = blockIdx.z;
  const int qbase = qt * 16;

  const __bf16* Qbh = Qh + ((size_t)(b * NH + h)) * SS * DH;
  const __bf16* Kbh = Kh + ((size_t)(b * NH + h)) * SS * DH;
  const __bf16* Vbh = Vt + ((size_t)(b * NH + h)) * DH * SS;

  // Q fragments for the K-chain over D=64 (two 16x16x32 steps), reused 16x
  v16bf aq[2];
#pragma unroll
  for (int j = 0; j < 2; ++j)
    aq[j] = make_frag(Qbh + (size_t)(qbase + mr) * DH + j * 32 + kh * 8);

  // ---- scores: each wave owns 16 of the 64 column tiles ----
  for (int ct = wave; ct < SS / 16; ct += 4) {
    v8f c = zero8();
    const int kcol = ct * 16;
#pragma unroll
    for (int j = 0; j < 2; ++j) {
      v16bf bk = make_frag(Kbh + (size_t)(kcol + mr) * DH + j * 32 + kh * 8);
      c = __builtin_amdgcn_wmma_f32_16x16x32_bf16(
          false, aq[j], false, bk, (short)0, c, false, false);
    }
#pragma unroll
    for (int r = 0; r < 8; ++r) {
      const int ml = r + 8 * kh;                 // local row 0..15
      const int ig = qbase + ml, jg = kcol + mr; // global q,k index
      float s = c[r] * 0.03125f;                 // 1/sqrt(1024)
      if (jg > ig) s = -1e9f;                    // causal mask (== triu k=1)
      sc[ml * SS + jg] = s;
    }
  }
  __syncthreads();

  // ---- softmax: each wave owns 4 rows; wave32 shuffle reductions ----
  for (int row = wave * 4; row < wave * 4 + 4; ++row) {
    float* srow = sc + row * SS;
    float mx = -3.4e38f;
    for (int j = lane; j < SS; j += 32) mx = fmaxf(mx, srow[j]);
#pragma unroll
    for (int o = 16; o > 0; o >>= 1) mx = fmaxf(mx, __shfl_xor(mx, o, 32));
    float sum = 0.0f;
    for (int j = lane; j < SS; j += 32) {
      float e = __expf(srow[j] - mx);
      srow[j] = e;
      sum += e;
    }
#pragma unroll
    for (int o = 16; o > 0; o >>= 1) sum += __shfl_xor(sum, o, 32);
    const float inv = 1.0f / sum;
    float* arow = attn + (((size_t)(b * NH + h)) * SS + qbase + row) * SS;
    for (int j = lane; j < SS; j += 32) {
      float p = srow[j] * inv;
      arow[j] = p;                    // attn_weights, written exactly once
      P[row * SS + j] = (__bf16)p;    // bf16 probs for PV matmul
    }
  }
  __syncthreads();

  // ---- PV: wave owns one 16-wide d tile; K-chain of 1024 = 32 WMMAs ----
  v8f c = zero8();
  const int dcol = wave * 16 + mr;
  for (int ks = 0; ks < 32; ++ks) {
    const int kk = ks * 32 + kh * 8;
    v16bf a  = make_frag(&P[mr * SS + kk]);
    v16bf bb = make_frag(Vbh + (size_t)dcol * SS + kk);
    c = __builtin_amdgcn_wmma_f32_16x16x32_bf16(
        false, a, false, bb, (short)0, c, false, false);
  }
#pragma unroll
  for (int r = 0; r < 8; ++r) {
    const int ml = r + 8 * kh;
    const int s  = qbase + ml;
    ctx[((size_t)(b * SS + s)) * DMODEL + h * DH + dcol] = (__bf16)c[r];
  }
}

// ---- host launcher -------------------------------------------------------
extern "C" void kernel_launch(void* const* d_in, const int* in_sizes, int n_in,
                              void* d_out, int out_size, void* d_ws, size_t ws_size,
                              hipStream_t stream) {
  (void)in_sizes; (void)n_in; (void)out_size; (void)ws_size;

  const float* q  = (const float*)d_in[0];
  const float* k  = (const float*)d_in[1];
  const float* v  = (const float*)d_in[2];
  const float* Wq = (const float*)d_in[3];
  const float* bq = (const float*)d_in[4];
  const float* Wk = (const float*)d_in[5];
  const float* bk = (const float*)d_in[6];
  const float* Wv = (const float*)d_in[7];
  const float* bv = (const float*)d_in[8];
  const float* Wo = (const float*)d_in[9];
  const float* bo = (const float*)d_in[10];
  // d_in[11] = causal_mask: applied analytically (triu, k=1)

  float* out = (float*)d_out;
  char*  ws  = (char*)d_ws;
  __bf16* Qh  = (__bf16*)(ws + WS_Q);
  __bf16* Kh  = (__bf16*)(ws + WS_K);
  __bf16* Vt  = (__bf16*)(ws + WS_VT);
  __bf16* ctx = (__bf16*)(ws + WS_CTX);

  float* x_out = out + OUT_X;
  float* presK = out + OUT_PRESK;
  float* presV = out + OUT_PRESV;
  float* attn  = out + OUT_ATTN;

  dim3 ggrid(DMODEL / 64, MROWS / 64);   // (16, 64)
  dim3 gblk(128);

  mha_gemm_kernel<<<ggrid, gblk, 0, stream>>>(q, Wq, bq, 0, Qh, (float*)nullptr);
  mha_gemm_kernel<<<ggrid, gblk, 0, stream>>>(k, Wk, bk, 1, Kh, presK);
  mha_gemm_kernel<<<ggrid, gblk, 0, stream>>>(v, Wv, bv, 2, Vt, presV);

  dim3 agrid(SS / 16, NH, BB);           // (64, 16, 4)
  const size_t attn_lds = (size_t)16 * SS * 4 + (size_t)16 * SS * 2; // 96 KB
  mha_attn_kernel<<<agrid, gblk, attn_lds, stream>>>(Qh, Kh, Vt, attn, ctx);

  mha_gemm_kernel<<<ggrid, gblk, 0, stream>>>(ctx, Wo, bo, 3, (__bf16*)nullptr, x_out);
}
